// AGNN_86878598463695
// MI455X (gfx1250) — matrix operand order repaired and checked by reading, not compiled, not run on hardware
//
#include <hip/hip_runtime.h>
#include <hip/hip_bf16.h>

typedef __attribute__((ext_vector_type(16))) _Float16 v16h;
typedef __attribute__((ext_vector_type(8)))  float    v8f;

#define DIN   75
#define NH    16
#define KP    96          // K padded to 3 x 32 for wmma_f32_16x16x32_f16

// ---------- helpers ----------
__device__ __forceinline__ float dot4f(float4 a, float4 b) {
    return a.x * b.x + a.y * b.y + a.z * b.z + a.w * b.w;
}
// order-preserving f32 -> u32 map (for atomicMax-based segment max)
__device__ __forceinline__ unsigned mapF32(float x) {
    unsigned u = __float_as_uint(x);
    return (u & 0x80000000u) ? ~u : (u | 0x80000000u);
}
__device__ __forceinline__ float unmapF32(unsigned m) {
    unsigned u = (m & 0x80000000u) ? (m & 0x7FFFFFFFu) : ~m;
    return __uint_as_float(u);
}

// ---------- edge prep: int64 -> int32, append self-loops ----------
__global__ void k_prep_edges(const long long* __restrict__ eidx,
                             int* __restrict__ srcI, int* __restrict__ dstI,
                             int E, int Etot) {
    int e = blockIdx.x * blockDim.x + threadIdx.x;
    if (e >= Etot) return;
    if (e < E) {
        srcI[e] = (int)eidx[e];
        dstI[e] = (int)eidx[(long long)E + e];
    } else {
        int n = e - E;          // self loop
        srcI[e] = n;
        dstI[e] = n;
    }
}

// ---------- fold lin2+pool+gather: v = W2^T * gw (16), c1 = b2·gw ----------
__global__ void k_prep_v(const float* __restrict__ w2, const float* __restrict__ b2,
                         const float* __restrict__ gw, float* __restrict__ vout) {
    __shared__ float sv[17];
    int t = threadIdx.x;
    if (t < 17) sv[t] = 0.0f;
    __syncthreads();
    if (t < 64) {
        float g = gw[t];
        #pragma unroll
        for (int k = 0; k < 16; ++k) atomicAdd(&sv[k], g * w2[t * 16 + k]);
        atomicAdd(&sv[16], g * b2[t]);
    }
    __syncthreads();
    if (t < 17) vout[t] = sv[t];
}

// ---------- lin1 + ReLU via V_WMMA_F32_16X16X32_F16 ----------
// 8 waves per block; each wave computes a 16-node x 16-channel tile.
// A/B fragments are built branch-free from zero-padded f16 LDS staging;
// both sides read two 8-half runs, 16 apart in K -> ds_load_b128 pairs.
__global__ void __launch_bounds__(256)
k_lin1_wmma(const float* __restrict__ x,
            const float* __restrict__ w1,
            const float* __restrict__ b1,
            float* __restrict__ h0, int N) {
    __shared__ _Float16 sW[NH * KP];        // [o][k] = W1[o][k], f16, zero-padded
    __shared__ _Float16 sX[8][16 * KP];     // per-wave A tile: [node][k], zero-padded

    const int t     = threadIdx.x;
    const int wv    = t >> 5;
    const int lane  = t & 31;
    const int tiles = N >> 4;
    int tile = (int)blockIdx.x * 8 + wv;
    if (tile >= tiles) tile = tiles - 1;    // clamp: EXEC stays full, barrier-safe

    // stage weights (block-cooperative), f32 -> f16, keep [o][k] layout, pad K
    for (int idx = t; idx < NH * KP; idx += 256) {
        int o = idx / KP, k = idx - o * KP;
        sW[idx] = (k < DIN) ? (_Float16)w1[o * DIN + k] : (_Float16)0.0f;
    }
    // stage this wave's 16-node x-tile with a coalesced linear stream
    _Float16* sx = sX[wv];
    const float* xb = x + (size_t)tile * 16 * DIN;
    for (int idx = lane; idx < 16 * DIN; idx += 32) {
        int node = idx / DIN, k = idx - node * DIN;
        sx[node * KP + k] = (_Float16)xb[idx];
    }
    for (int idx = lane; idx < 16 * (KP - DIN); idx += 32) {   // zero the pad
        int node = idx / (KP - DIN), k = DIN + (idx - node * (KP - DIN));
        sx[node * KP + k] = (_Float16)0.0f;
    }
    __syncthreads();

    const int hi = lane >> 4;
    const int m  = lane & 15;               // node-in-tile (A) / channel (B,D)
    v8f acc = {};
    #pragma unroll
    for (int k0 = 0; k0 < KP; k0 += 32) {
        v16h a, b;
        #pragma unroll
        for (int i = 0; i < 16; ++i) {
            // A (16x32 f16): slots 0..7 -> K=k0+hi*8+0..7, slots 8..15 -> +16
            int ka = k0 + ((i < 8) ? 0 : 16) + hi * 8 + (i & 7);
            a[i] = sx[m * KP + ka];
            // B (32x16 f16): slots 0..7 -> K=k0+hi*8+0..7, slots 8..15 -> +16; col=m
            int kb = k0 + ((i < 8) ? 0 : 16) + hi * 8 + (i & 7);
            b[i] = sW[m * KP + kb];
        }
        acc = __builtin_amdgcn_wmma_f32_16x16x32_f16(
                  false, a, false, b, (short)0, acc, false, false);
    }
    float bias = b1[m];
    #pragma unroll
    for (int j = 0; j < 8; ++j) {
        int node = tile * 16 + j + 8 * hi;  // D row = j + 8*(lane>=16)
        float val = acc[j] + bias;
        h0[node * NH + m] = val > 0.0f ? val : 0.0f;
    }
}

// ---------- per-node: L2-normalize features, clear accumulators ----------
__global__ void k_norm_clear(const float* __restrict__ h,
                             float* __restrict__ hn,
                             float* __restrict__ acc,
                             unsigned* __restrict__ mmax,
                             float* __restrict__ ssum, int N) {
    int i = blockIdx.x * blockDim.x + threadIdx.x;
    if (i >= N) return;
    const float4* hv = (const float4*)(h + (size_t)i * NH);
    float4 r0 = hv[0], r1 = hv[1], r2 = hv[2], r3 = hv[3];
    float ss = dot4f(r0, r0) + dot4f(r1, r1) + dot4f(r2, r2) + dot4f(r3, r3);
    float inv = 1.0f / fmaxf(sqrtf(ss), 1e-12f);
    float4* ov = (float4*)(hn + (size_t)i * NH);
    ov[0] = make_float4(r0.x * inv, r0.y * inv, r0.z * inv, r0.w * inv);
    ov[1] = make_float4(r1.x * inv, r1.y * inv, r1.z * inv, r1.w * inv);
    ov[2] = make_float4(r2.x * inv, r2.y * inv, r2.z * inv, r2.w * inv);
    ov[3] = make_float4(r3.x * inv, r3.y * inv, r3.z * inv, r3.w * inv);
    float4 z = make_float4(0.f, 0.f, 0.f, 0.f);
    float4* av = (float4*)(acc + (size_t)i * NH);
    av[0] = z; av[1] = z; av[2] = z; av[3] = z;
    mmax[i] = 0u;          // mapped "-inf" (every node has a self-loop, so always updated)
    ssum[i] = 0.0f;
}

// ---------- edge pass 1: alpha = beta * cos(src,dst); segment max ----------
__global__ void k_edge_alpha(const float* __restrict__ hn,
                             const int* __restrict__ src, const int* __restrict__ dst,
                             const float* __restrict__ beta,
                             float* __restrict__ alpha,
                             unsigned* __restrict__ mmax, int Etot) {
    int e = blockIdx.x * blockDim.x + threadIdx.x;
    if (e >= Etot) return;
    __builtin_prefetch(src + e + 8192, 0, 0);   // global_prefetch_b8: stream-ahead
    __builtin_prefetch(dst + e + 8192, 0, 0);
    int s = src[e], d = dst[e];
    const float4* a4 = (const float4*)(hn + (size_t)s * NH);
    const float4* b4 = (const float4*)(hn + (size_t)d * NH);
    float dt = dot4f(a4[0], b4[0]) + dot4f(a4[1], b4[1]) +
               dot4f(a4[2], b4[2]) + dot4f(a4[3], b4[3]);
    float al = beta[0] * dt;
    alpha[e] = al;
    atomicMax(&mmax[d], mapF32(al));
}

// ---------- edge pass 2: ex = exp(alpha - max); segment sum ----------
__global__ void k_edge_expsum(float* __restrict__ alpha,
                              const int* __restrict__ dst,
                              const unsigned* __restrict__ mmax,
                              float* __restrict__ ssum, int Etot) {
    int e = blockIdx.x * blockDim.x + threadIdx.x;
    if (e >= Etot) return;
    int d = dst[e];
    float ex = __expf(alpha[e] - unmapF32(mmax[d]));
    alpha[e] = ex;
    atomicAdd(&ssum[d], ex);
}

// ---------- edge pass 3: out[dst] += h[src] * (ex / sum[dst]) ----------
__global__ void k_edge_scatter(const float* __restrict__ h,
                               const int* __restrict__ src, const int* __restrict__ dst,
                               const float* __restrict__ alpha,
                               const float* __restrict__ ssum,
                               float* __restrict__ acc, int Etot) {
    int e = blockIdx.x * blockDim.x + threadIdx.x;
    if (e >= Etot) return;
    __builtin_prefetch(src + e + 8192, 0, 0);
    __builtin_prefetch(alpha + e + 8192, 0, 0);
    int s = src[e], d = dst[e];
    float w = alpha[e] / ssum[d];
    const float4* hv = (const float4*)(h + (size_t)s * NH);
    float* o = acc + (size_t)d * NH;
    #pragma unroll
    for (int q = 0; q < 4; ++q) {
        float4 r = hv[q];
        atomicAdd(o + q * 4 + 0, r.x * w);
        atomicAdd(o + q * 4 + 1, r.y * w);
        atomicAdd(o + q * 4 + 2, r.z * w);
        atomicAdd(o + q * 4 + 3, r.w * w);
    }
}

// ---------- readout ----------
__global__ void k_init_out(float* __restrict__ out, const float* __restrict__ gb, int G) {
    int g = blockIdx.x * blockDim.x + threadIdx.x;
    if (g < G) out[g] = gb[0];
}

__global__ void k_final(const float* __restrict__ h,
                        const long long* __restrict__ batch,
                        const float* __restrict__ vc,   // v[0..15], c1 at [16]
                        float* __restrict__ out, int N) {
    int i = blockIdx.x * blockDim.x + threadIdx.x;
    if (i >= N) return;
    const float4* hv = (const float4*)(h + (size_t)i * NH);
    const float4* vv = (const float4*)vc;
    float t = dot4f(hv[0], vv[0]) + dot4f(hv[1], vv[1]) +
              dot4f(hv[2], vv[2]) + dot4f(hv[3], vv[3]) + vc[16];
    atomicAdd(out + (int)batch[i], t);
}

extern "C" void kernel_launch(void* const* d_in, const int* in_sizes, int n_in,
                              void* d_out, int out_size, void* d_ws, size_t ws_size,
                              hipStream_t stream) {
    const float*     x     = (const float*)d_in[0];
    const long long* eidx  = (const long long*)d_in[1];
    const long long* batch = (const long long*)d_in[2];
    // d_in[3] = num_graphs (device scalar) -> use out_size instead
    const float* w1    = (const float*)d_in[4];
    const float* b1    = (const float*)d_in[5];
    const float* beta1 = (const float*)d_in[6];
    const float* beta2 = (const float*)d_in[7];
    const float* w2    = (const float*)d_in[8];
    const float* b2    = (const float*)d_in[9];
    const float* gw    = (const float*)d_in[10];
    const float* gb    = (const float*)d_in[11];
    float* out = (float*)d_out;

    const int N    = in_sizes[0] / DIN;
    const int E    = in_sizes[1] / 2;
    const int G    = out_size;
    const int Etot = E + N;

    // carve workspace (all regions fully rewritten every call -> deterministic)
    char* ws = (char*)d_ws;
    size_t off = 0;
    auto take = [&](size_t bytes) -> char* {
        char* p = ws + off;
        off = (off + bytes + 255) & ~(size_t)255;
        return p;
    };
    float*    h0   = (float*)take((size_t)N * NH * 4);
    float*    hA   = (float*)take((size_t)N * NH * 4);   // normalized features
    float*    hB   = (float*)take((size_t)N * NH * 4);
    unsigned* mm   = (unsigned*)take((size_t)N * 4);
    float*    sb   = (float*)take((size_t)N * 4);
    float*    al   = (float*)take((size_t)Etot * 4);
    int*      srcI = (int*)take((size_t)Etot * 4);
    int*      dstI = (int*)take((size_t)Etot * 4);
    float*    vc   = (float*)take(256);
    (void)ws_size; (void)n_in;

    const int TB  = 256;                   // 8 wave32 per block
    const int ebl = (Etot + TB - 1) / TB;
    const int nbl = (N + TB - 1) / TB;

    // prep (independent of convs)
    k_prep_edges<<<ebl, TB, 0, stream>>>(eidx, srcI, dstI, E, Etot);
    k_prep_v<<<1, 64, 0, stream>>>(w2, b2, gw, vc);

    // lin1 + ReLU (WMMA): 8 tiles of 16 nodes per block
    const int tiles = N / 16;
    k_lin1_wmma<<<(tiles + 7) / 8, TB, 0, stream>>>(x, w1, b1, h0, N);

    // conv1: h0 -> hB
    k_norm_clear<<<nbl, TB, 0, stream>>>(h0, hA, hB, mm, sb, N);
    k_edge_alpha<<<ebl, TB, 0, stream>>>(hA, srcI, dstI, beta1, al, mm, Etot);
    k_edge_expsum<<<ebl, TB, 0, stream>>>(al, dstI, mm, sb, Etot);
    k_edge_scatter<<<ebl, TB, 0, stream>>>(h0, srcI, dstI, al, sb, hB, Etot);

    // conv2: hB -> h0
    k_norm_clear<<<nbl, TB, 0, stream>>>(hB, hA, h0, mm, sb, N);
    k_edge_alpha<<<ebl, TB, 0, stream>>>(hA, srcI, dstI, beta2, al, mm, Etot);
    k_edge_expsum<<<ebl, TB, 0, stream>>>(al, dstI, mm, sb, Etot);
    k_edge_scatter<<<ebl, TB, 0, stream>>>(hB, srcI, dstI, al, sb, h0, Etot);

    // folded lin2 + global_add_pool + gather head
    k_init_out<<<(G + TB - 1) / TB, TB, 0, stream>>>(out, gb, G);
    k_final<<<nbl, TB, 0, stream>>>(h0, batch, vc, out, N);
}
